// MultiHeadAttention_39891656245282
// MI455X (gfx1250) — matrix operand order repaired and checked by reading, not compiled, not run on hardware
//
#include <hip/hip_runtime.h>
#include <hip/hip_bf16.h>

// ---------------------------------------------------------------------------
// Multi-head attention, MI455X (gfx1250), bf16 WMMA pipeline.
//   B=2, S=2048, D=1024, H=16, DK=64, scale = 1/sqrt(D) = 1/32.
//   1. x (fp32) -> x_bf (bf16)
//   2. W{q,k,v,o} (fp32 [K][N]) -> Wt (bf16 [N][K])
//   3. GEMM (WMMA bf16, async-LDS double-buffered): Q,K=[B,H,S,DK], Vt=[B,H,DK,S]
//   4. flash attention (WMMA QK^T, fp32 online softmax, WMMA P*V) -> ctx bf16
//   5. GEMM: out fp32 = ctx @ Wo + bo
// Workspace: ~48 MB.
// ---------------------------------------------------------------------------

#define B_SZ   2
#define S_LEN  2048
#define D_MOD  1024
#define NH     16
#define DK     64
#define SCALE  0.03125f   // 1/sqrt(1024)

#if __has_builtin(__builtin_amdgcn_global_load_async_to_lds_b128) && \
    __has_builtin(__builtin_amdgcn_s_wait_asynccnt)
#define HAS_ASYNC 1
#else
#define HAS_ASYNC 0
#endif

typedef __attribute__((ext_vector_type(16))) __bf16 v16bf;
typedef __attribute__((ext_vector_type(8)))  float  v8f;

struct Pack32 { uint4 a, b; };

static __device__ __forceinline__ v16bf make_frag(uint4 x, uint4 y) {
    Pack32 p{x, y};
    return __builtin_bit_cast(v16bf, p);
}

static __device__ __forceinline__ unsigned short f2bf(float f) {
    __hip_bfloat16 h = __float2bfloat16(f);   // RNE
    return __builtin_bit_cast(unsigned short, h);
}

#if HAS_ASYNC
// 16B per lane, global -> LDS, tracked by ASYNCcnt (no VGPR data path).
// Builtin signature (from clang diagnostic): param0 = <4 x i32> __device__*
typedef int v4i __attribute__((vector_size(16)));
typedef __attribute__((address_space(1))) v4i* gas_v4i;
typedef __attribute__((address_space(3))) v4i* las_v4i;

static __device__ __forceinline__ void async_cp16(unsigned short* lds,
                                                  const unsigned short* g) {
    __builtin_amdgcn_global_load_async_to_lds_b128(
        (gas_v4i)(v4i*)const_cast<unsigned short*>(g),
        (las_v4i)(v4i*)lds,
        0 /*imm offset*/, 0 /*cpol*/);
}
#endif

// ---------------------------------------------------------------------------
// 1) fp32 -> bf16 elementwise
// ---------------------------------------------------------------------------
__global__ void k_f32_to_bf16(const float* __restrict__ in,
                              unsigned short* __restrict__ out, int n) {
    int i = blockIdx.x * 256 + threadIdx.x;
    if (i < n) out[i] = f2bf(in[i]);
}

// ---------------------------------------------------------------------------
// 2) fp32 [R][C] -> bf16 [C][R] transpose+convert (R=C=1024)
// ---------------------------------------------------------------------------
__global__ __launch_bounds__(256)
void k_transpose_bf16(const float* __restrict__ in,
                      unsigned short* __restrict__ out, int R, int C) {
    __shared__ float tile[32][33];
    int x  = blockIdx.x * 32 + threadIdx.x;
    int y0 = blockIdx.y * 32 + threadIdx.y;
#pragma unroll
    for (int j = 0; j < 4; ++j)
        tile[threadIdx.y + j * 8][threadIdx.x] = in[(size_t)(y0 + j * 8) * C + x];
    __syncthreads();
    int xo  = blockIdx.y * 32 + threadIdx.x;
    int yo0 = blockIdx.x * 32 + threadIdx.y;
#pragma unroll
    for (int j = 0; j < 4; ++j)
        out[(size_t)(yo0 + j * 8) * R + xo] = f2bf(tile[threadIdx.x][threadIdx.y + j * 8]);
}

// ---------------------------------------------------------------------------
// 3/5) bf16 GEMM:  C[M x N] = A[M x Kd] * Bt[N x Kd]^T + bias
//   256 threads (8 waves), C tile 128x128, wave tile 32x64 (2x4 WMMA),
//   K-step 32, double-buffered LDS panels, async global->LDS when available.
//   mode 0: bf16 out scatter [B,H,S,DK]; mode 1: bf16 [B,H,DK,S]; mode 2: fp32 [M][N]
// ---------------------------------------------------------------------------
__global__ __launch_bounds__(256)
void k_gemm_bf16(const unsigned short* __restrict__ A,
                 const unsigned short* __restrict__ Bt,
                 const float* __restrict__ bias,
                 void* __restrict__ out,
                 int M, int N, int Kd, int mode) {
    __shared__ __align__(16) unsigned short Als[2][128 * 32];
    __shared__ __align__(16) unsigned short Bls[2][128 * 32];

    const int tid  = threadIdx.x;
    const int wave = tid >> 5;
    const int lane = tid & 31;
    const int lo   = lane & 15, hi = lane >> 4;
    const int wm   = wave >> 1, wn = wave & 1;       // 4 x 2 wave grid
    const int m0   = blockIdx.y * 128;
    const int n0   = blockIdx.x * 128;
    const int ku4  = Kd >> 3;                        // row stride (uint4)

    // stage one 128x32 panel pair (512 uint4 each) into buffer `buf`
    auto stage = [&](int buf, int k0) {
#pragma unroll
        for (int j = 0; j < 2; ++j) {
            int i = tid + 256 * j;        // 0..511
            int row = i >> 2, c = i & 3;  // 4 uint4 per row
#if HAS_ASYNC
            async_cp16(&Als[buf][(size_t)i * 8],
                       A + (size_t)(m0 + row) * Kd + k0 + c * 8);
            async_cp16(&Bls[buf][(size_t)i * 8],
                       Bt + (size_t)(n0 + row) * Kd + k0 + c * 8);
#else
            reinterpret_cast<uint4*>(Als[buf])[i] =
                reinterpret_cast<const uint4*>(A)[(size_t)(m0 + row) * ku4 + (k0 >> 3) + c];
            reinterpret_cast<uint4*>(Bls[buf])[i] =
                reinterpret_cast<const uint4*>(Bt)[(size_t)(n0 + row) * ku4 + (k0 >> 3) + c];
#endif
        }
    };

    v8f acc[2][4] = {};
    const int nk = Kd >> 5;

#if HAS_ASYNC
    stage(0, 0);
#endif
    for (int it = 0; it < nk; ++it) {
#if HAS_ASYNC
        __builtin_amdgcn_s_wait_asynccnt(0);
        __syncthreads();
        if (it + 1 < nk) stage((it + 1) & 1, (it + 1) << 5);  // overlap with WMMA
        const uint4* Al = reinterpret_cast<const uint4*>(Als[it & 1]);
        const uint4* Bl = reinterpret_cast<const uint4*>(Bls[it & 1]);
#else
        __syncthreads();
        stage(0, it << 5);
        __syncthreads();
        const uint4* Al = reinterpret_cast<const uint4*>(Als[0]);
        const uint4* Bl = reinterpret_cast<const uint4*>(Bls[0]);
#endif
        v16bf afr[2], bfr[4];
#pragma unroll
        for (int i = 0; i < 2; ++i) {
            int r = wm * 32 + i * 16 + lo;            // A: lane = row M
            afr[i] = make_frag(Al[r * 4 + hi], Al[r * 4 + 2 + hi]);
        }
#pragma unroll
        for (int j = 0; j < 4; ++j) {
            int r = wn * 64 + j * 16 + lo;            // B: lane = col N
            bfr[j] = make_frag(Bl[r * 4 + hi], Bl[r * 4 + 2 + hi]);
        }
#pragma unroll
        for (int i = 0; i < 2; ++i)
#pragma unroll
            for (int j = 0; j < 4; ++j)
                acc[i][j] = __builtin_amdgcn_wmma_f32_16x16x32_bf16(
                    false, afr[i], false, bfr[j], (short)0, acc[i][j], false, false);
    }

    // Epilogue: C layout vgpr v -> M = v + hi*8 ; N = lo
#pragma unroll
    for (int i = 0; i < 2; ++i)
#pragma unroll
        for (int j = 0; j < 4; ++j)
#pragma unroll
            for (int v = 0; v < 8; ++v) {
                int row = m0 + wm * 32 + i * 16 + v + hi * 8;
                int col = n0 + wn * 64 + j * 16 + lo;
                float val = acc[i][j][v] + bias[col];
                if (mode == 2) {
                    ((float*)out)[(size_t)row * N + col] = val;
                } else {
                    int b  = row >> 11;     // / 2048
                    int s  = row & 2047;
                    int h  = col >> 6;      // / 64
                    int dk = col & 63;
                    unsigned short bv = f2bf(val);
                    if (mode == 0)          // [B,H,S,DK]
                        ((unsigned short*)out)
                            [(((size_t)b * NH + h) * S_LEN + s) * DK + dk] = bv;
                    else                    // [B,H,DK,S]
                        ((unsigned short*)out)
                            [(((size_t)b * NH + h) * DK + dk) * S_LEN + s] = bv;
                }
            }
}

// ---------------------------------------------------------------------------
// 4) Flash attention. grid (S/64, H, B), 128 threads = 4 waves.
//    wave w owns q rows [qt*64 + w*16, +16). Causal mask == tril(mask).
//    K/V tiles double-buffered, async-staged; softmax in fp32.
// ---------------------------------------------------------------------------
__global__ __launch_bounds__(128)
void k_flash_attn(const unsigned short* __restrict__ Q,
                  const unsigned short* __restrict__ K,
                  const unsigned short* __restrict__ Vt,
                  unsigned short* __restrict__ ctx) {
    __shared__ __align__(16) unsigned short Kls[2][64 * 64];   // [key][dk]
    __shared__ __align__(16) unsigned short Vls[2][64 * 64];   // [dk][key]
    __shared__ __align__(16) unsigned short Pls[4 * 16 * 64];

    const int qt   = blockIdx.x;
    const int h    = blockIdx.y;
    const int b    = blockIdx.z;
    const int tid  = threadIdx.x;
    const int wave = tid >> 5;
    const int lane = tid & 31;
    const int lo   = lane & 15, hi = lane >> 4;

    const size_t bh = (size_t)(b * NH + h);
    const unsigned short* Qb = Q  + bh * S_LEN * DK;
    const unsigned short* Kb = K  + bh * S_LEN * DK;
    const unsigned short* Vb = Vt + bh * DK * S_LEN;
    const int qrow0 = qt * 64 + wave * 16;

    auto stageKV = [&](int buf, int kt) {
#pragma unroll
        for (int j = 0; j < 4; ++j) {
            int i = tid + 128 * j;          // 0..511
            int row = i >> 3, c = i & 7;    // 8 uint4 per row of 64 elems
#if HAS_ASYNC
            async_cp16(&Kls[buf][(size_t)i * 8],
                       Kb + (size_t)(kt * 64 + row) * DK + c * 8);
            async_cp16(&Vls[buf][(size_t)i * 8],
                       Vb + (size_t)row * S_LEN + (size_t)kt * 64 + c * 8);
#else
            reinterpret_cast<uint4*>(Kls[buf])[i] =
                reinterpret_cast<const uint4*>(Kb)[(size_t)(kt * 64 + row) * 8 + c];
            reinterpret_cast<uint4*>(Vls[buf])[i] =
                reinterpret_cast<const uint4*>(Vb)[(size_t)row * (S_LEN / 8) + (size_t)kt * 8 + c];
#endif
        }
    };

    // Q fragments (A layout), resident for the whole k loop.
    v16bf qfr[2];
    {
        const uint4* Qu = reinterpret_cast<const uint4*>(Qb);
        int r = qrow0 + lo;                       // row stride = 8 uint4
#pragma unroll
        for (int kc = 0; kc < 2; ++kc)
            qfr[kc] = make_frag(Qu[(size_t)r * 8 + kc * 4 + hi],
                                Qu[(size_t)r * 8 + kc * 4 + 2 + hi]);
    }

    v8f   oacc[4] = {};
    float m_run[8], l_run[8];
#pragma unroll
    for (int j = 0; j < 8; ++j) { m_run[j] = -3.0e38f; l_run[j] = 0.f; }

    const int nkt = qt + 1;
#if HAS_ASYNC
    stageKV(0, 0);
#endif
    for (int kt = 0; kt < nkt; ++kt) {
#if HAS_ASYNC
        __builtin_amdgcn_s_wait_asynccnt(0);
        __syncthreads();
        if (kt + 1 < nkt) stageKV((kt + 1) & 1, kt + 1);   // overlap with compute
        const unsigned short* Kc = Kls[kt & 1];
        const unsigned short* Vc = Vls[kt & 1];
#else
        __syncthreads();
        stageKV(0, kt);
        __syncthreads();
        const unsigned short* Kc = Kls[0];
        const unsigned short* Vc = Vls[0];
#endif

        // scores: 16 q x 64 k  (4 N-subtiles x 2 dk-chunks)
        v8f sc[4];
        const uint4* Klu = reinterpret_cast<const uint4*>(Kc);
#pragma unroll
        for (int n = 0; n < 4; ++n) {
            v8f a = {};
#pragma unroll
            for (int kc = 0; kc < 2; ++kc) {
                int kr = n * 16 + lo;            // B: lane = key col
                v16bf bfr = make_frag(Klu[kr * 8 + kc * 4 + hi],
                                      Klu[kr * 8 + kc * 4 + 2 + hi]);
                a = __builtin_amdgcn_wmma_f32_16x16x32_bf16(
                        false, qfr[kc], false, bfr, (short)0, a, false, false);
            }
            sc[n] = a;
        }

        // scale + causal mask + per-tile row max
        const int kbase = kt * 64;
        float mtile[8];
#pragma unroll
        for (int j = 0; j < 8; ++j) {
            int q = qrow0 + j + hi * 8;
            float mj = -3.0e38f;
#pragma unroll
            for (int n = 0; n < 4; ++n) {
                int k = kbase + n * 16 + lo;
                float v = sc[n][j] * SCALE;
                v = (k <= q) ? v : -3.0e38f;
                sc[n][j] = v;
                mj = fmaxf(mj, v);
            }
            for (int m = 1; m < 16; m <<= 1)
                mj = fmaxf(mj, __shfl_xor(mj, m, 32));
            mtile[j] = mj;
        }

        // online softmax update
        float scale_o[8];
#pragma unroll
        for (int j = 0; j < 8; ++j) {
            float mnew = fmaxf(m_run[j], mtile[j]);
            scale_o[j] = __expf(m_run[j] - mnew);
            m_run[j]   = mnew;
        }
#pragma unroll
        for (int j = 0; j < 8; ++j) {
            float s = 0.f;
#pragma unroll
            for (int n = 0; n < 4; ++n) {
                float p = __expf(sc[n][j] - m_run[j]);
                sc[n][j] = p;
                s += p;
            }
            for (int m = 1; m < 16; m <<= 1) s += __shfl_xor(s, m, 32);
            l_run[j] = l_run[j] * scale_o[j] + s;
        }
#pragma unroll
        for (int d = 0; d < 4; ++d)
#pragma unroll
            for (int j = 0; j < 8; ++j) oacc[d][j] *= scale_o[j];

        // P: C layout -> per-wave LDS scratch [16][64] (same-wave DS ordering)
        unsigned short* Pw = Pls + wave * 16 * 64;
#pragma unroll
        for (int n = 0; n < 4; ++n)
#pragma unroll
            for (int j = 0; j < 8; ++j)
                Pw[(j + hi * 8) * 64 + n * 16 + lo] = f2bf(sc[n][j]);

        // reload P as A fragments, accumulate O += P * V
        const uint4* Pu = reinterpret_cast<const uint4*>(Pw);
        v16bf pfr[2];
#pragma unroll
        for (int kc = 0; kc < 2; ++kc)
            pfr[kc] = make_frag(Pu[lo * 8 + kc * 4 + hi],
                                Pu[lo * 8 + kc * 4 + 2 + hi]);

        const uint4* Vlu = reinterpret_cast<const uint4*>(Vc);
#pragma unroll
        for (int d = 0; d < 4; ++d) {
            v8f a = oacc[d];
#pragma unroll
            for (int kc = 0; kc < 2; ++kc) {
                int vr = d * 16 + lo;            // B: lane = dk col
                v16bf vfr = make_frag(Vlu[vr * 8 + kc * 4 + hi],
                                      Vlu[vr * 8 + kc * 4 + 2 + hi]);
                a = __builtin_amdgcn_wmma_f32_16x16x32_bf16(
                        false, pfr[kc], false, vfr, (short)0, a, false, false);
            }
            oacc[d] = a;
        }
    }

    // normalize and store ctx bf16 [B][S][D]
    float inv_l[8];
#pragma unroll
    for (int j = 0; j < 8; ++j) inv_l[j] = 1.0f / l_run[j];
    unsigned short* Cb = ctx + (size_t)b * S_LEN * D_MOD;
#pragma unroll
    for (int d = 0; d < 4; ++d)
#pragma unroll
        for (int j = 0; j < 8; ++j) {
            int q   = qrow0 + j + hi * 8;
            int col = h * 64 + d * 16 + lo;
            Cb[(size_t)q * D_MOD + col] = f2bf(oacc[d][j] * inv_l[j]);
        }
}

// ---------------------------------------------------------------------------
// launch
// ---------------------------------------------------------------------------
extern "C" void kernel_launch(void* const* d_in, const int* in_sizes, int n_in,
                              void* d_out, int out_size, void* d_ws, size_t ws_size,
                              hipStream_t stream) {
    (void)in_sizes; (void)n_in; (void)out_size; (void)ws_size;

    const float* x  = (const float*)d_in[0];
    // d_in[1] = mask (tril) -> implemented as causal predicate in-kernel
    const float* Wq = (const float*)d_in[2];
    const float* bq = (const float*)d_in[3];
    const float* Wk = (const float*)d_in[4];
    const float* bk = (const float*)d_in[5];
    const float* Wv = (const float*)d_in[6];
    const float* bv = (const float*)d_in[7];
    const float* Wo = (const float*)d_in[8];
    const float* bo = (const float*)d_in[9];
    float* out = (float*)d_out;

    char* w = (char*)d_ws;
    const size_t MB = 1024ull * 1024ull;
    unsigned short* x_bf   = (unsigned short*)(w + 0 * MB);   // 8 MB
    unsigned short* wq_t   = (unsigned short*)(w + 8 * MB);   // 2 MB
    unsigned short* wk_t   = (unsigned short*)(w + 10 * MB);  // 2 MB
    unsigned short* wv_t   = (unsigned short*)(w + 12 * MB);  // 2 MB
    unsigned short* wo_t   = (unsigned short*)(w + 14 * MB);  // 2 MB
    unsigned short* q_bf   = (unsigned short*)(w + 16 * MB);  // 8 MB
    unsigned short* k_bf   = (unsigned short*)(w + 24 * MB);  // 8 MB
    unsigned short* vt_bf  = (unsigned short*)(w + 32 * MB);  // 8 MB
    unsigned short* ctx_bf = (unsigned short*)(w + 40 * MB);  // 8 MB

    const int M = B_SZ * S_LEN;      // 4096
    const int N = D_MOD;             // 1024
    const int Kd = D_MOD;            // 1024
    const int nx = M * D_MOD;        // 4,194,304

    k_f32_to_bf16<<<(nx + 255) / 256, 256, 0, stream>>>(x, x_bf, nx);

    dim3 tg(32, 32), tb(32, 8);
    k_transpose_bf16<<<tg, tb, 0, stream>>>(Wq, wq_t, Kd, N);
    k_transpose_bf16<<<tg, tb, 0, stream>>>(Wk, wk_t, Kd, N);
    k_transpose_bf16<<<tg, tb, 0, stream>>>(Wv, wv_t, Kd, N);
    k_transpose_bf16<<<tg, tb, 0, stream>>>(Wo, wo_t, Kd, N);

    dim3 gg(N / 128, M / 128);       // (8, 32)
    k_gemm_bf16<<<gg, 256, 0, stream>>>(x_bf, wq_t, bq, q_bf,  M, N, Kd, 0);
    k_gemm_bf16<<<gg, 256, 0, stream>>>(x_bf, wk_t, bk, k_bf,  M, N, Kd, 0);
    k_gemm_bf16<<<gg, 256, 0, stream>>>(x_bf, wv_t, bv, vt_bf, M, N, Kd, 1);

    dim3 fg(S_LEN / 64, NH, B_SZ);   // (32, 16, 2)
    k_flash_attn<<<fg, 128, 0, stream>>>(q_bf, k_bf, vt_bf, ctx_bf);

    k_gemm_bf16<<<gg, 256, 0, stream>>>(ctx_bf, wo_t, bo, (void*)out, M, N, Kd, 2);
}
